// _ConvCaps_79267916415228
// MI455X (gfx1250) — matrix operand order repaired and checked by reading, not compiled, not run on hardware
//
#include <hip/hip_runtime.h>
#include <hip/hip_bf16.h>
#include <math.h>

// ---------------- problem constants (from reference) ----------------
#define BSZ      8
#define H_IN     16
#define W_IN     16
#define KK       3
#define STRIDE   2
#define OH       7
#define OW       7
#define IN_CAPS  32
#define OUT_CAPS 32
#define P_SIZE   16
#define C_IN     544            // 32*17
#define NK       288            // K*K*IN_CAPS
#define NPIX     392            // B*OH*OW
#define NK4      72             // NK/4 k-groups
#define N_TILE   576            // NK4 * (OUT_CAPS/4)
#define LN2PI_H  1.8378770664093453f   // ln(2*pi); need 16 * LN2PI/2 = 8*LN2PI

// WMMA vector types (CDNA5, wave32)
typedef __attribute__((ext_vector_type(16))) __bf16 v16bf;
typedef __attribute__((ext_vector_type(8)))  float  v8f;

union BF16Vec { v16bf v; unsigned short s[16]; };

__device__ __forceinline__ unsigned short f32_to_bf16(float f) {
    union { float f; unsigned u; } x; x.f = f;
    unsigned u = x.u;
    u += 0x7FFFu + ((u >> 16) & 1u);     // round-to-nearest-even
    return (unsigned short)(u >> 16);
}
__device__ __forceinline__ float bf16_to_f32(unsigned short h) {
    union { float f; unsigned u; } x; x.u = ((unsigned)h) << 16;
    return x.f;
}
__device__ __forceinline__ unsigned pack_bf16x2(float lo, float hi) {
    return (unsigned)f32_to_bf16(lo) | ((unsigned)f32_to_bf16(hi) << 16);
}

// --------------------------------------------------------------------
// Kernel 0: pre-swizzle weights (f32 -> bf16) into the exact per-lane
// B-operand layout used here, same k order as A (32x16 bf16: lanes 0-15 K=0..15,
// lanes 16-31 K=16..31, N = lane&15). Contraction slot t = (k_sub<<2)|j,
// only k_sub<4 is live (A is block-diagonal over 4 k's); k_sub 4..7 = 0.
// 576 tiles * 512 bf16 = 576 KiB in d_ws; shared by all 392 positions.
// --------------------------------------------------------------------
__global__ void _ConvCaps_wswz_kernel(const float* __restrict__ W,
                                      unsigned short* __restrict__ wswz) {
    int idx = blockIdx.x * blockDim.x + threadIdx.x;   // 0 .. 294911
    if (idx >= N_TILE * 512) return;
    int tt   = idx >> 9;          // tile = k4*8 + o4
    int li   = idx & 511;
    int lane = li >> 4;
    int e    = li & 15;
    int Kt   = ((lane & 16) ? 16 : 0) + e;   // contraction row 0..31
    int k2   = Kt >> 2, j = Kt & 3;
    int Nn   = lane & 15;
    int o_sub = Nn >> 2, m = Nn & 3;
    int k4 = tt >> 3, o4 = tt & 7;
    float val = 0.0f;
    if (k2 < 4) {
        int k = k4 * 4 + k2;                  // < 288
        int o = o4 * 4 + o_sub;               // < 32
        val = W[((k * OUT_CAPS + o) * 4 + j) * 4 + m];   // weights[k][o][j][m]
    }
    wswz[idx] = f32_to_bf16(val);
}

// --------------------------------------------------------------------
// Kernel 1: fully fused ConvCaps + EM routing.
// One workgroup (256 threads = 8 waves) per spatial position n.
// Votes (288 x 32 x 16) live in LDS as bf16 (288 KB, needs CDNA5's
// 320 KB/WGP) — computed ONCE via v_wmma_f32_16x16x32_bf16, then all
// 3 M-steps + 2 E-steps are LDS-resident (zero HBM re-reads).
// In-LDS vote layout uses p' = m*4+i so each lane's 4 D-rows per k are
// address-consecutive -> 2x ds_store_b64 per WMMA.
// --------------------------------------------------------------------
extern __shared__ unsigned char smem_raw[];

#define SMEM_BYTES 322176

__global__ void __launch_bounds__(256)
_ConvCaps_em_kernel(const float* __restrict__ x,
                    const unsigned short* __restrict__ wswz,
                    const float* __restrict__ beta_u,
                    const float* __restrict__ beta_a,
                    float* __restrict__ out) {
    // ---- LDS layout ----
    unsigned short* votes = (unsigned short*)smem_raw;         // 288*512 bf16
    unsigned short* r_lds = votes + NK * 512;                  // 288*32  bf16
    float* fbase  = (float*)(r_lds + NK * OUT_CAPS);
    float* acts   = fbase;            // 288
    float* mean_l = acts + NK;        // 512  (o*16 + p')
    float* lnstd  = mean_l + 512;     // 512
    float* inv2v  = lnstd + 512;      // 512
    float* rsum   = inv2v + 512;      // 32
    float* invrs  = rsum + 32;        // 32
    float* lnact  = invrs + 32;       // 32
    float* actout = lnact + 32;       // 32
    float* scratch = actout + 32;     // 256 (8 partials x 32 o)

    const int tid  = threadIdx.x;
    const int wave = tid >> 5;
    const int lane = tid & 31;

    const int n  = blockIdx.x;            // 0..391
    const int b  = n / (OH * OW);
    const int rm = n % (OH * OW);
    const int oy = rm / OW;
    const int ox = rm % OW;

    // ---- Phase A: stage activations + init r = 1/32 (vectorized) ----
    for (int kk = tid; kk < NK; kk += 256) {
        int kr = kk / 96, kc = (kk / 32) % 3, cap = kk & 31;
        acts[kk] = x[((b * H_IN + (oy * STRIDE + kr)) * W_IN + (ox * STRIDE + kc)) * C_IN
                     + IN_CAPS * P_SIZE + cap];
    }
    {
        unsigned ri = pack_bf16x2(1.0f / OUT_CAPS, 1.0f / OUT_CAPS);
        uint4 ri4 = make_uint4(ri, ri, ri, ri);
        uint4* r4 = (uint4*)r_lds;
        for (int idx = tid; idx < NK * OUT_CAPS / 8; idx += 256) r4[idx] = ri4;
    }

    // ---- Phase B: votes via WMMA ----
    // Outer loop over k-groups (9 per wave): A built once per 8 WMMAs.
    // Inner loop over o-groups: B tiles are a contiguous 8 KB block.
    for (int k4 = wave; k4 < NK4; k4 += 8) {
        const int k_base = k4 * 4;

        // A operand: block-diagonal poses. Row M = lane&15 -> (k1,i).
        const int M  = lane & 15;
        const int k1 = M >> 2, i = M & 3;
        const int kk = k_base + k1;
        const int kr = kk / 96, kc = (kk / 32) % 3, cap = kk & 31;
        const float4 p4 = *(const float4*)&x[
            ((b * H_IN + (oy * STRIDE + kr)) * W_IN + (ox * STRIDE + kc)) * C_IN
            + cap * P_SIZE + i * 4];

        BF16Vec a;
        #pragma unroll
        for (int e = 0; e < 16; ++e) {
            int g = e >> 1;
            // A lane layout matches B: lanes 0-15 K=0..15, lanes 16-31 K=16..31
            int Kt = ((lane & 16) ? 16 : 0) + e;
            int k2 = Kt >> 2, j = Kt & 3;
            float pv = (j == 0) ? p4.x : (j == 1) ? p4.y : (j == 2) ? p4.z : p4.w;
            a.s[e] = (k2 == k1) ? f32_to_bf16(pv) : (unsigned short)0;
        }

        // Prefetch next k-group's 8 KB of B tiles (gfx1250 global_prefetch).
        if (k4 + 8 < NK4)
            __builtin_prefetch((const void*)(wswz + (size_t)(k4 + 8) * 8 * 512
                                             + lane * 128), 0, 1);

        const int Nn = lane & 15;
        const int o_sub = Nn >> 2, m = Nn & 3;
        const int k1a = (lane < 16) ? 0 : 2;      // D rows this lane holds

        for (int o4 = 0; o4 < 8; ++o4) {
            // B operand: pre-swizzled weights, 32 contiguous bytes per lane.
            BF16Vec bb;
            const uint4* wp = (const uint4*)(wswz + (size_t)(k4 * 8 + o4) * 512
                                             + lane * 16);
            ((uint4*)bb.s)[0] = wp[0];
            ((uint4*)bb.s)[1] = wp[1];

            v8f c = {};
            c = __builtin_amdgcn_wmma_f32_16x16x32_bf16(
                    /*neg_a=*/false, a.v, /*neg_b=*/false, bb.v,
                    /*c_mod=*/(short)0, c, /*reuse_a=*/false, /*reuse_b=*/false);

            // D rows rr=0..3 -> (k1a, id=0..3), rr=4..7 -> (k1a+1, id=0..3).
            // With p' = m*4 + i the 4 ids are consecutive bf16 -> b64 stores.
            unsigned short* d0 = votes + (k_base + k1a    ) * 512
                                       + (o4 * 4 + o_sub) * 16 + m * 4;
            unsigned short* d1 = votes + (k_base + k1a + 1) * 512
                                       + (o4 * 4 + o_sub) * 16 + m * 4;
            uint2 w0, w1;
            w0.x = pack_bf16x2(c[0], c[1]); w0.y = pack_bf16x2(c[2], c[3]);
            w1.x = pack_bf16x2(c[4], c[5]); w1.y = pack_bf16x2(c[6], c[7]);
            *(uint2*)d0 = w0;
            *(uint2*)d1 = w1;
        }
    }
    __syncthreads();

    // ---- Phase C: EM routing (3 iterations), all LDS-resident ----
    float lam = 0.001f;
    for (int t = 0; t < 3; ++t) {
        lam += 0.0001f;

        // M-step (1): rr = r*act, normalize over o (b128 row access).
        for (int kk = tid; kk < NK; kk += 256) {
            float a_in = acts[kk];
            uint4* rrow = (uint4*)(r_lds + kk * OUT_CAPS);
            uint4 rq[4];
            #pragma unroll
            for (int q = 0; q < 4; ++q) rq[q] = rrow[q];
            unsigned* w = (unsigned*)rq;
            float tmp[OUT_CAPS];
            float s = 0.0f;
            #pragma unroll
            for (int o = 0; o < OUT_CAPS; ++o) {
                unsigned u = w[o >> 1];
                unsigned short h = (o & 1) ? (unsigned short)(u >> 16)
                                           : (unsigned short)(u & 0xffff);
                float rv = bf16_to_f32(h) * a_in;
                tmp[o] = rv; s += rv;
            }
            float inv = 1.0f / (s + 1e-8f);
            #pragma unroll
            for (int q = 0; q < 16; ++q)
                w[q] = pack_bf16x2(tmp[2 * q] * inv, tmp[2 * q + 1] * inv);
            #pragma unroll
            for (int q = 0; q < 4; ++q) rrow[q] = rq[q];
        }
        __syncthreads();

        // M-step (2): rsum[o] = sum_k rr[k][o]  (8-way tree in LDS)
        {
            int o = tid & 31, part = tid >> 5;          // 8 partials per o
            float s = 0.0f;
            for (int kk = part * 36; kk < part * 36 + 36; ++kk)
                s += bf16_to_f32(r_lds[kk * OUT_CAPS + o]);
            scratch[part * 32 + o] = s;
        }
        __syncthreads();
        if (tid < 32) {
            float s = 0.0f;
            #pragma unroll
            for (int p = 0; p < 8; ++p) s += scratch[p * 32 + tid];
            rsum[tid]  = s;
            invrs[tid] = 1.0f / (s + 1e-8f);
        }
        __syncthreads();

        // M-step (3): mean / E[v^2] per (o,p'); std^2 = E[v^2]-mean^2.
        for (int op = tid; op < 512; op += 256) {
            int o = op >> 4;
            float irs = invrs[o];
            float m1 = 0.0f, m2 = 0.0f;
            for (int kk = 0; kk < NK; ++kk) {
                float rp = bf16_to_f32(r_lds[kk * OUT_CAPS + o]) * irs;
                float v  = bf16_to_f32(votes[kk * 512 + op]);
                m1 += rp * v;
                m2 += rp * v * v;
            }
            float var = m2 - m1 * m1;
            var = (var < 0.0f) ? 0.0f : var;
            float sd = sqrtf(var + 1e-8f);
            mean_l[op] = m1;
            lnstd[op]  = __logf(sd);
            inv2v[op]  = 0.5f / (var + 1e-8f);
        }
        __syncthreads();

        // M-step (4): cost + activation.
        if (tid < 32) {
            float bu = beta_u[tid];
            float c = 0.0f;
            #pragma unroll
            for (int p = 0; p < P_SIZE; ++p) c += bu + lnstd[tid * 16 + p];
            c *= rsum[tid];
            float ao = 1.0f / (1.0f + __expf(-lam * (beta_a[tid] - c)));
            actout[tid] = ao;
            lnact[tid]  = __logf(ao + 1e-30f);
        }
        __syncthreads();

        // E-step: r[k][o] = softmax_o( sum_p ln_p + ln act_out )
        if (t < 2) {
            for (int kk = tid; kk < NK; kk += 256) {
                float lp[OUT_CAPS];
                float mx = -1e30f;
                #pragma unroll
                for (int o = 0; o < OUT_CAPS; ++o) {
                    const uint4* vrow = (const uint4*)(votes + kk * 512 + o * 16);
                    uint4 va = vrow[0], vb = vrow[1];
                    unsigned vw[8] = { va.x, va.y, va.z, va.w,
                                       vb.x, vb.y, vb.z, vb.w };
                    float s = lnact[o] - 8.0f * LN2PI_H;
                    #pragma unroll
                    for (int q = 0; q < 8; ++q) {
                        unsigned u = vw[q];
                        int p0 = 2 * q, p1 = 2 * q + 1;
                        float v0 = bf16_to_f32((unsigned short)(u & 0xffff));
                        float v1 = bf16_to_f32((unsigned short)(u >> 16));
                        float d0 = v0 - mean_l[o * 16 + p0];
                        float d1 = v1 - mean_l[o * 16 + p1];
                        s -= d0 * d0 * inv2v[o * 16 + p0] + lnstd[o * 16 + p0];
                        s -= d1 * d1 * inv2v[o * 16 + p1] + lnstd[o * 16 + p1];
                    }
                    lp[o] = s;
                    mx = (s > mx) ? s : mx;
                }
                float den = 0.0f;
                #pragma unroll
                for (int o = 0; o < OUT_CAPS; ++o) {
                    float e_ = __expf(lp[o] - mx);
                    lp[o] = e_; den += e_;
                }
                float inv = 1.0f / den;
                uint4 rq[4];
                unsigned* w = (unsigned*)rq;
                #pragma unroll
                for (int q = 0; q < 16; ++q)
                    w[q] = pack_bf16x2(lp[2 * q] * inv, lp[2 * q + 1] * inv);
                uint4* rrow = (uint4*)(r_lds + kk * OUT_CAPS);
                #pragma unroll
                for (int q = 0; q < 4; ++q) rrow[q] = rq[q];
            }
            __syncthreads();
        }
    }

    // ---- Phase D: write output. Internal p' = m*4+i -> external p = i*4+m.
    float* outp = out + (size_t)n * C_IN;   // 544 floats per position
    for (int opt = tid; opt < 512; opt += 256) {
        int o = opt >> 4, pt = opt & 15;
        int i2 = pt >> 2, m2 = pt & 3;
        outp[opt] = mean_l[o * 16 + m2 * 4 + i2];
    }
    if (tid < 32) outp[512 + tid] = actout[tid];
}

// --------------------------------------------------------------------
extern "C" void kernel_launch(void* const* d_in, const int* in_sizes, int n_in,
                              void* d_out, int out_size, void* d_ws, size_t ws_size,
                              hipStream_t stream) {
    (void)in_sizes; (void)n_in; (void)out_size; (void)ws_size;

    const float* x      = (const float*)d_in[0];
    const float* W      = (const float*)d_in[1];
    const float* beta_u = (const float*)d_in[2];
    const float* beta_a = (const float*)d_in[3];
    float* out          = (float*)d_out;
    unsigned short* wswz = (unsigned short*)d_ws;   // needs 589,824 bytes

    // Raise dynamic-LDS cap to use CDNA5's 320 KB/WGP (idempotent, capture-safe).
    hipFuncSetAttribute((const void*)_ConvCaps_em_kernel,
                        hipFuncAttributeMaxDynamicSharedMemorySize, SMEM_BYTES);

    _ConvCaps_wswz_kernel<<<(N_TILE * 512 + 255) / 256, 256, 0, stream>>>(W, wswz);
    _ConvCaps_em_kernel<<<NPIX, 256, SMEM_BYTES, stream>>>(x, wswz, beta_u, beta_a, out);
}
